// Set_60696477827724
// MI455X (gfx1250) — matrix operand order repaired and checked by reading, not compiled, not run on hardware
//
#include <hip/hip_runtime.h>

// ---------------- problem constants (match reference) ----------------
#define NTOT   262144   // total ragged elements
#define NB     4096     // number of segments
#define DIM    512      // feature dim D (= H*QS = H*ES)
#define NH     8        // heads
#define CHUNK  64       // rows of a segment processed per pass (4 row-tiles)
#define ESTRIDE 520     // bf16 elements per LDS row (pad vs 512 to dodge bank conflicts)
#define PSTRIDE 516     // pooled LDS row stride (floats)

typedef __bf16 bf16;
typedef __attribute__((ext_vector_type(16))) __bf16 v16bf;
typedef __attribute__((ext_vector_type(8)))  float  v8f;
typedef __attribute__((ext_vector_type(4)))  unsigned int u32x4;
typedef __attribute__((ext_vector_type(4)))  float  f32x4;

union Frag {           // 16 bf16 = 32 bytes = 2 x 128-bit loads
    v16bf v;
    u32x4 q[2];
};

// ------------- prep kernel 1: fp32 [D, 512] -> bf16 [512, D] (transposed) -------------
__global__ __launch_bounds__(256) void prep_weights(
        const float* __restrict__ Wq, const float* __restrict__ Wk,
        const float* __restrict__ Wv,
        bf16* __restrict__ Wq_t, bf16* __restrict__ Wk_t, bf16* __restrict__ Wv_t) {
    int i = blockIdx.x * 256 + threadIdx.x;        // 0 .. 3*512*512-1
    int which = i >> 18;
    int rem   = i & ((1 << 18) - 1);
    int o = rem >> 9;          // output column 0..511
    int d = rem & 511;         // input dim 0..511
    const float* src = (which == 0) ? Wq : ((which == 1) ? Wk : Wv);
    bf16* dst        = (which == 0) ? Wq_t : ((which == 1) ? Wk_t : Wv_t);
    dst[(size_t)o * DIM + d] = (bf16)src[(size_t)d * DIM + o];
}

// ------------- prep kernel 2: segment offsets via binary search (ids sorted) -------------
__global__ __launch_bounds__(256) void prep_offsets(const int* __restrict__ seg,
                                                    int* __restrict__ off) {
    int b = blockIdx.x * 256 + threadIdx.x;
    if (b > NB) return;
    int lo = 0, hi = NTOT;                 // lower_bound(seg, b)
    while (lo < hi) {
        int mid = (lo + hi) >> 1;
        if (seg[mid] < b) lo = mid + 1; else hi = mid;
    }
    off[b] = lo;
}

// ------------- main fused kernel: one workgroup (8 waves) per segment -------------
__global__ __launch_bounds__(256, 2) void seg_attn(
        const float* __restrict__ enc,
        const bf16* __restrict__ Wq_t, const bf16* __restrict__ Wk_t,
        const bf16* __restrict__ Wv_t,
        const float* __restrict__ bq, const float* __restrict__ bk,
        const float* __restrict__ bv,
        const int* __restrict__ off,
        float* __restrict__ out) {
    __shared__ __align__(16) bf16 encA[CHUNK * ESTRIDE];   // chunk rows, bf16 (~66.5KB)
    __shared__ float scoreW[CHUNK][NH];                    // scores -> exp weights
    __shared__ float pooled[NH * PSTRIDE];                 // running attn-weighted enc
    __shared__ float mS[NH], dS[NH], scS[NH];

    const int b    = blockIdx.x;
    const int t    = threadIdx.x;
    const int lane = t & 31;
    const int wave = t >> 5;

    const int segStart  = off[b];
    const int rowsTotal = off[b + 1] - segStart;

    for (int i = t; i < NH * PSTRIDE; i += 256) pooled[i] = 0.f;
    if (t < NH) { mS[t] = -3.0e38f; dS[t] = 0.f; }
    __syncthreads();

    for (int base = 0; base < rowsTotal; base += CHUNK) {
        const int rc = min(CHUNK, rowsTotal - base);

        // ---- stage chunk into LDS as bf16 (zero-pad missing rows) ----
        for (int g = t; g < CHUNK * (DIM / 4); g += 256) {
            int r  = g / (DIM / 4);
            int c4 = (g % (DIM / 4)) * 4;
            f32x4 val = (f32x4)(0.f);
            if (r < rc)
                val = *(const f32x4*)(enc + (size_t)(segStart + base + r) * DIM + c4);
            bf16* dst = &encA[r * ESTRIDE + c4];
            dst[0] = (bf16)val.x; dst[1] = (bf16)val.y;
            dst[2] = (bf16)val.z; dst[3] = (bf16)val.w;
        }
        // prefetch next chunk while we compute (2 x 256 lanes x 256B = 128KB chunk)
        if (base + CHUNK < rowsTotal) {
            const float* np = enc + (size_t)(segStart + base + CHUNK) * DIM;
            __builtin_prefetch(np + t * 64, 0, 0);
            __builtin_prefetch(np + 16384 + t * 64, 0, 0);
        }
        __syncthreads();

        // ---- Q/K projection via WMMA; per-row score = diag(Q K^T) per head ----
        // One HEAD per wave; each wave computes all FOUR 16-row tiles so every
        // weight (B) fragment loaded from L2 feeds four WMMAs. The head's 64
        // columns are split into two sequential passes (jp) to bound register
        // pressure; the pass loop must NOT be unrolled or both accumulator
        // sets go live at once and spill.
        const int h     = wave;                    // head 0..7
        const int koff  = (lane < 16) ? 0 : 8;     // A-frag K sub-offset
        const int mloc  = lane & 15;
        const int ncol  = lane & 15;
        const v8f vzero = {0.f, 0.f, 0.f, 0.f, 0.f, 0.f, 0.f, 0.f};

        float p[4][8];                             // per row-tile score partials
#pragma unroll
        for (int rt = 0; rt < 4; ++rt)
#pragma unroll
            for (int v = 0; v < 8; ++v) p[rt][v] = 0.f;

#pragma clang loop unroll(disable)
        for (int jp = 0; jp < 2; ++jp) {           // column-pair pass (NOT unrolled)
            v8f qacc[4][2], kacc[4][2];            // [row-tile][j2]
#pragma unroll
            for (int rt = 0; rt < 4; ++rt) {
                qacc[rt][0] = vzero; qacc[rt][1] = vzero;
                kacc[rt][0] = vzero; kacc[rt][1] = vzero;
            }

#pragma clang loop unroll_count(2)
            for (int kb = 0; kb < DIM; kb += 32) {
                Frag af[4];
#pragma unroll
                for (int rt = 0; rt < 4; ++rt) {
                    const int mrow = rt * 16 + mloc;
                    af[rt].q[0] = *(const u32x4*)(&encA[mrow * ESTRIDE + kb + koff]);
                    af[rt].q[1] = *(const u32x4*)(&encA[mrow * ESTRIDE + kb + 16 + koff]);
                }
                const int kw = kb + ((lane < 16) ? 0 : 8);  // B-frag K base
#pragma unroll
                for (int j2 = 0; j2 < 2; ++j2) {
                    const int col = h * 64 + (jp * 2 + j2) * 16 + ncol;
                    Frag bqf, bkf;
                    const bf16* wqp = Wq_t + (size_t)col * DIM + kw;
                    bqf.q[0] = *(const u32x4*)wqp;
                    bqf.q[1] = *(const u32x4*)(wqp + 16);
                    const bf16* wkp = Wk_t + (size_t)col * DIM + kw;
                    bkf.q[0] = *(const u32x4*)wkp;
                    bkf.q[1] = *(const u32x4*)(wkp + 16);
#pragma unroll
                    for (int rt = 0; rt < 4; ++rt) {
                        qacc[rt][j2] = __builtin_amdgcn_wmma_f32_16x16x32_bf16(
                            false, af[rt].v, false, bqf.v, (short)0, qacc[rt][j2], false, false);
                        kacc[rt][j2] = __builtin_amdgcn_wmma_f32_16x16x32_bf16(
                            false, af[rt].v, false, bkf.v, (short)0, kacc[rt][j2], false, false);
                    }
                }
            }

            // fold this pass into score partials (add biases, elementwise product)
#pragma unroll
            for (int j2 = 0; j2 < 2; ++j2) {
                const int col = h * 64 + (jp * 2 + j2) * 16 + ncol;
                const float bqv = bq[col], bkv = bk[col];
#pragma unroll
                for (int rt = 0; rt < 4; ++rt)
#pragma unroll
                    for (int v = 0; v < 8; ++v)
                        p[rt][v] += (qacc[rt][j2][v] + bqv) * (kacc[rt][j2][v] + bkv);
            }
        }

        // reduce across the 16 lanes that hold the head's columns, emit scores
#pragma unroll
        for (int rt = 0; rt < 4; ++rt) {
#pragma unroll
            for (int v = 0; v < 8; ++v) {
                float x = p[rt][v];
                x += __shfl_xor(x, 1, 32);
                x += __shfl_xor(x, 2, 32);
                x += __shfl_xor(x, 4, 32);
                x += __shfl_xor(x, 8, 32);
                p[rt][v] = x;
            }
            if ((lane & 15) == 0) {
                const int rbase = rt * 16 + ((lane >> 4) << 3);  // +8 for upper half
#pragma unroll
                for (int v = 0; v < 8; ++v)
                    scoreW[rbase + v][h] = p[rt][v] * 0.125f;   // 1/sqrt(QS)
            }
        }
        __syncthreads();

        // ---- online softmax state update (one thread per head) ----
        if (t < NH) {
            float cm = -3.0e38f;
            for (int n = 0; n < rc; ++n) cm = fmaxf(cm, scoreW[n][t]);
            const float m0 = mS[t];
            const float nm = fmaxf(m0, cm);
            const float scale = (m0 == -3.0e38f) ? 0.f : __expf(m0 - nm);
            float wsum = 0.f;
            for (int n = 0; n < rc; ++n) {
                float w = __expf(scoreW[n][t] - nm);
                scoreW[n][t] = w;                 // reuse as exp-weights
                wsum += w;
            }
            dS[t]  = dS[t] * scale + wsum;
            mS[t]  = nm;
            scS[t] = scale;
        }
        __syncthreads();

        // ---- pooled[h][d] = pooled*scale + sum_n w[n,h]*enc[n,d] ----
        for (int d = t; d < DIM; d += 256) {
            float acc[NH];
#pragma unroll
            for (int hh = 0; hh < NH; ++hh) acc[hh] = 0.f;
            for (int n = 0; n < rc; ++n) {
                const float e = (float)encA[n * ESTRIDE + d];
#pragma unroll
                for (int hh = 0; hh < NH; ++hh) acc[hh] += scoreW[n][hh] * e;
            }
#pragma unroll
            for (int hh = 0; hh < NH; ++hh)
                pooled[hh * PSTRIDE + d] = pooled[hh * PSTRIDE + d] * scS[hh] + acc[hh];
        }
        __syncthreads();
    }

    // ---- out[b, o] = (pooled[h] . Wv[:, o]) / denom[h] + bv[o] ----
    for (int o = t; o < DIM; o += 256) {
        const int h = o >> 6;
        const bf16* wv = Wv_t + (size_t)o * DIM;
        float dot = 0.f;
        for (int d = 0; d < DIM; d += 8) {
            u32x4 raw = *(const u32x4*)(wv + d);
            const bf16* w8 = (const bf16*)&raw;
#pragma unroll
            for (int i = 0; i < 8; ++i)
                dot += pooled[h * PSTRIDE + d + i] * (float)w8[i];
        }
        float res = 0.f;
        if (rowsTotal > 0) res = dot / dS[h] + bv[o];
        out[(size_t)b * DIM + o] = res;
    }
}

// ---------------------------------------------------------------------
extern "C" void kernel_launch(void* const* d_in, const int* in_sizes, int n_in,
                              void* d_out, int out_size, void* d_ws, size_t ws_size,
                              hipStream_t stream) {
    const float* enc = (const float*)d_in[0];
    const int*   seg = (const int*)  d_in[1];
    const float* Wq  = (const float*)d_in[2];
    const float* bq  = (const float*)d_in[3];
    const float* Wk  = (const float*)d_in[4];
    const float* bk  = (const float*)d_in[5];
    const float* Wv  = (const float*)d_in[6];
    const float* bv  = (const float*)d_in[7];
    float* out = (float*)d_out;

    char* ws = (char*)d_ws;
    bf16* Wq_t = (bf16*)(ws);
    bf16* Wk_t = (bf16*)(ws + (size_t)DIM * DIM * 2);
    bf16* Wv_t = (bf16*)(ws + (size_t)2 * DIM * DIM * 2);
    int*  offp = (int*) (ws + (size_t)3 * DIM * DIM * 2);

    prep_weights<<<(3 * DIM * DIM) / 256, 256, 0, stream>>>(Wq, Wk, Wv, Wq_t, Wk_t, Wv_t);
    prep_offsets<<<(NB + 1 + 255) / 256, 256, 0, stream>>>(seg, offp);
    seg_attn<<<NB, 256, 0, stream>>>(enc, Wq_t, Wk_t, Wv_t, bq, bk, bv, offp, out);
}
